// SDPAttention_43224550867179
// MI455X (gfx1250) — matrix operand order repaired and checked by reading, not compiled, not run on hardware
//
#include <hip/hip_runtime.h>

#define EMBED   1024
#define HEADS   16
#define HD      64
#define BATCH   4
#define SEQ     2048
#define NTOK    (BATCH * SEQ)        // 8192
#define THREE_D (3 * EMBED)          // 3072
#define LDS_STRIDE 34                // P tile: 32 + 2 pad (4B-aligned pair loads)
#define KSTRIDE 72                   // K tile: 64 + 8 pad (conflict-free, 16B-aligned)
#define VSTRIDE 40                   // V tile: 32 + 8 pad (conflict-free, 16B-aligned)

typedef __bf16 bf16_t;
typedef __attribute__((ext_vector_type(2)))  __bf16 v2bf;
typedef __attribute__((ext_vector_type(8)))  __bf16 v8bf;
typedef __attribute__((ext_vector_type(16))) __bf16 v16bf;
typedef __attribute__((ext_vector_type(8)))  float  v8f;

__device__ __forceinline__ v8f wmma_bf16(v16bf a, v16bf b, v8f c) {
  return __builtin_amdgcn_wmma_f32_16x16x32_bf16(false, a, false, b, (short)0, c,
                                                 false, false);
}

// Async global -> LDS copy of 16 bytes (per lane), tracked by ASYNCcnt.
// VDST operand carries the LDS byte offset (flat LDS aperture addr[31:0]).
__device__ __forceinline__ void async_ld_b128(bf16_t* lds_dst, const bf16_t* gsrc) {
  const unsigned loff = (unsigned)(unsigned long long)lds_dst;
  asm volatile("global_load_async_to_lds_b128 %0, %1, off"
               :
               : "v"(loff), "v"(gsrc)
               : "memory");
}
__device__ __forceinline__ void wait_asynccnt0() {
  asm volatile("s_wait_asynccnt 0x0" ::: "memory");
}

// A fragment (16x32 bf16): row m = lane%16; VGPR v holds K pair
// klo = 2v, high half-lanes add +16 to K.
__device__ __forceinline__ v16bf load_a_frag(const bf16_t* rowbase, int k0, int lane) {
  const int hi8 = (lane >> 4) << 4;
  v16bf a;
#pragma unroll
  for (int v = 0; v < 8; ++v) {
    const int klo = 2 * v;
    const v2bf p = *reinterpret_cast<const v2bf*>(rowbase + k0 + klo + hi8);
    a[2 * v]     = p.x;
    a[2 * v + 1] = p.y;
  }
  return a;
}

// B fragment (32x16 bf16), B[k][n] taken from a row-major [N][K] source:
// lane holds column n = lane%16; elements e=0..15 are contiguous K,
// high half-lanes add +16 to K. Caller passes &src[n*ldk + kbase(+hi16)].
__device__ __forceinline__ v16bf load_b_frag(const bf16_t* p) {
  const v8bf b0 = *reinterpret_cast<const v8bf*>(p);
  const v8bf b1 = *reinterpret_cast<const v8bf*>(p + 8);
  v16bf b;
#pragma unroll
  for (int i = 0; i < 8; ++i) { b[i] = b0[i]; b[8 + i] = b1[i]; }
  return b;
}

// ---------------- fp32 -> bf16 convert ----------------
__global__ void cvt_bf16_kernel(const float* __restrict__ src,
                                bf16_t* __restrict__ dst, int n) {
  int i = blockIdx.x * blockDim.x + threadIdx.x;
  const int stride = gridDim.x * blockDim.x;
  for (; i < n; i += stride) dst[i] = (__bf16)src[i];
}

// ---------------- QKV projection ----------------
// C[32x64] per wave: qkv[n, o] = sum_d X[n,d] * Wqkv[o,d] + b[o]
// Epilogue scatters to Q,K [B,H,S,hd] and V transposed [B,H,hd,S], all bf16.
__global__ void qkv_gemm_kernel(const bf16_t* __restrict__ X,
                                const bf16_t* __restrict__ W,
                                const float* __restrict__ bias,
                                bf16_t* __restrict__ Qb,
                                bf16_t* __restrict__ Kb,
                                bf16_t* __restrict__ Vt) {
  const int lane = threadIdx.x & 31;
  const int wave = threadIdx.x >> 5;
  const int gw   = blockIdx.x * (blockDim.x >> 5) + wave;
  const int CT   = THREE_D / 64;                 // 48 col tiles
  const int rt   = gw / CT;                      // 0..255 (32-row tiles)
  const int ct   = gw % CT;
  if (rt >= NTOK / 32) return;

  const int m    = lane & 15;
  const int hi16 = (lane >> 4) << 4;
  const int hi8  = (lane >> 4) << 3;
  const int oc0  = ct * 64;

  const bf16_t* xrow0 = X + (size_t)(rt * 32 + m) * EMBED;       // A rows 0..15
  const bf16_t* xrow1 = xrow0 + (size_t)16 * EMBED;              // A rows 16..31

  v8f acc[2][4];
#pragma unroll
  for (int g = 0; g < 2; ++g)
#pragma unroll
    for (int j = 0; j < 4; ++j) acc[g][j] = (v8f){0.f,0.f,0.f,0.f,0.f,0.f,0.f,0.f};

  for (int k0 = 0; k0 < EMBED; k0 += 32) {
    __builtin_prefetch(xrow0 + k0 + 64, 0, 3);
    __builtin_prefetch(xrow1 + k0 + 64, 0, 3);
    const v16bf a0 = load_a_frag(xrow0, k0, lane);
    const v16bf a1 = load_a_frag(xrow1, k0, lane);
    v16bf b[4];
#pragma unroll
    for (int j = 0; j < 4; ++j)
      b[j] = load_b_frag(W + (size_t)(oc0 + j * 16 + m) * EMBED + k0 + hi16);
#pragma unroll
    for (int j = 0; j < 4; ++j) {
      acc[0][j] = wmma_bf16(a0, b[j], acc[0][j]);
      acc[1][j] = wmma_bf16(a1, b[j], acc[1][j]);
    }
  }

#pragma unroll
  for (int g = 0; g < 2; ++g)
#pragma unroll
    for (int j = 0; j < 4; ++j)
#pragma unroll
      for (int r = 0; r < 8; ++r) {
        const int tok = rt * 32 + g * 16 + r + hi8;  // C row
        const int bb  = tok >> 11;                   // / SEQ
        const int ss  = tok & (SEQ - 1);
        const int o   = oc0 + j * 16 + m;            // C col
        const float val = acc[g][j][r] + bias[o];
        const int three = o >> 10;
        const int h     = (o >> 6) & 15;
        const int e     = o & 63;
        const bf16_t bv = (__bf16)val;
        if (three == 0)
          Qb[(((bb * HEADS + h) * SEQ) + ss) * HD + e] = bv;
        else if (three == 1)
          Kb[(((bb * HEADS + h) * SEQ) + ss) * HD + e] = bv;
        else
          Vt[(((bb * HEADS + h) * HD) + e) * SEQ + ss] = bv;
      }
}

// ---------------- Flash attention (block-cooperative, async staging) ----------
// 8 waves of a block share one (b,h); each wave owns 16 query rows.
// Per 32-key iteration the block async-copies the K tile (32x64) and V tile
// (64x32, from transposed V) straight into LDS, waits ASYNCcnt, barriers,
// then all waves consume them.
__global__ void flash_attn_kernel(const bf16_t* __restrict__ Q,
                                  const bf16_t* __restrict__ K,
                                  const bf16_t* __restrict__ Vt,
                                  bf16_t* __restrict__ O) {
  __shared__ __align__(16) bf16_t ldsK[32 * KSTRIDE];
  __shared__ __align__(16) bf16_t ldsV[HD * VSTRIDE];
  __shared__ __align__(16) bf16_t ldsP[8][16 * LDS_STRIDE];

  const int tid  = threadIdx.x;
  const int lane = tid & 31;
  const int wave = tid >> 5;
  const int bh   = blockIdx.x >> 4;              // 0..63  (= b*HEADS + h)
  const int qt   = ((blockIdx.x & 15) << 3) + wave;  // 0..127
  const int h    = bh & (HEADS - 1);
  const int b    = bh >> 4;

  const int m    = lane & 15;
  const int hi16 = (lane >> 4) << 4;
  const int hi8  = (lane >> 4) << 3;

  const bf16_t* qbase = Q  + ((size_t)bh * SEQ + qt * 16) * HD;
  const bf16_t* kbase = K  + (size_t)bh * SEQ * HD;
  const bf16_t* vbase = Vt + (size_t)bh * HD * SEQ;
  bf16_t* myLds = &ldsP[wave][0];

  // staging thread mapping: one b128 per thread per tile
  const int krow = tid >> 3, kcol = (tid & 7) << 3;   // K tile 32x64
  const int vrow = tid >> 2, vcol = (tid & 3) << 3;   // V tile 64x32

  // Q A-fragments for d = 0..31 and 32..63 (held in registers)
  v16bf qa[2];
#pragma unroll
  for (int ks = 0; ks < 2; ++ks)
    qa[ks] = load_a_frag(qbase + (size_t)m * HD, ks * 32, lane);

  const float scale = 0.125f;                    // 1/sqrt(64)
  float mrow[8], lrow[8];
#pragma unroll
  for (int r = 0; r < 8; ++r) { mrow[r] = -INFINITY; lrow[r] = 0.f; }
  v8f acc[4];
#pragma unroll
  for (int j = 0; j < 4; ++j) acc[j] = (v8f){0.f,0.f,0.f,0.f,0.f,0.f,0.f,0.f};

  for (int kt = 0; kt < SEQ; kt += 32) {
    // ---- async staging of K/V tiles directly into LDS ----
    __builtin_prefetch(kbase + (size_t)(kt + 32 + krow) * HD + kcol, 0, 3);
    __builtin_prefetch(vbase + (size_t)vrow * SEQ + kt + 32 + vcol, 0, 3);
    async_ld_b128(ldsK + krow * KSTRIDE + kcol,
                  kbase + (size_t)(kt + krow) * HD + kcol);
    async_ld_b128(ldsV + vrow * VSTRIDE + vcol,
                  vbase + (size_t)vrow * SEQ + kt + vcol);
    wait_asynccnt0();
    __syncthreads();

    // scores: S[16q x 32k] = Q(16x64) . K^T, two 16-col halves
    v8f sf[2] = {{0.f,0.f,0.f,0.f,0.f,0.f,0.f,0.f},
                 {0.f,0.f,0.f,0.f,0.f,0.f,0.f,0.f}};
#pragma unroll
    for (int jj = 0; jj < 2; ++jj) {
      const int skey = jj * 16 + m;              // key row (within tile)
#pragma unroll
      for (int ks = 0; ks < 2; ++ks) {
        const v16bf kb = load_b_frag(ldsK + skey * KSTRIDE + ks * 32 + hi16);
        sf[jj] = wmma_bf16(qa[ks], kb, sf[jj]);
      }
    }
#pragma unroll
    for (int jj = 0; jj < 2; ++jj)
#pragma unroll
      for (int r = 0; r < 8; ++r) sf[jj][r] *= scale;

    // online softmax: row max across 16 lanes of each half-wave
    float alpha[8];
#pragma unroll
    for (int r = 0; r < 8; ++r) {
      float mx = fmaxf(sf[0][r], sf[1][r]);
#pragma unroll
      for (int off = 8; off > 0; off >>= 1)
        mx = fmaxf(mx, __shfl_xor(mx, off, 16));
      const float mnew = fmaxf(mrow[r], mx);
      alpha[r] = __expf(mrow[r] - mnew);
      mrow[r]  = mnew;
    }
#pragma unroll
    for (int jj = 0; jj < 2; ++jj)
#pragma unroll
      for (int r = 0; r < 8; ++r) sf[jj][r] = __expf(sf[jj][r] - mrow[r]);
#pragma unroll
    for (int r = 0; r < 8; ++r) {
      float rs = sf[0][r] + sf[1][r];
#pragma unroll
      for (int off = 8; off > 0; off >>= 1) rs += __shfl_xor(rs, off, 16);
      lrow[r] = lrow[r] * alpha[r] + rs;
    }
#pragma unroll
    for (int j = 0; j < 4; ++j)
#pragma unroll
      for (int r = 0; r < 8; ++r) acc[j][r] *= alpha[r];

    // relayout P (C-layout) -> A-layout via per-wave LDS tile
#pragma unroll
    for (int jj = 0; jj < 2; ++jj)
#pragma unroll
      for (int r = 0; r < 8; ++r)
        myLds[(r + hi8) * LDS_STRIDE + jj * 16 + m] = (__bf16)sf[jj][r];

    const v16bf pa = load_a_frag(myLds + m * LDS_STRIDE, 0, lane);

    // O += P(16x32) . V(32x64), V tile read from LDS [d][kk]
#pragma unroll
    for (int j = 0; j < 4; ++j) {
      const int d = j * 16 + m;                  // output-col for this lane
      const v16bf vb = load_b_frag(ldsV + d * VSTRIDE + hi16);
      acc[j] = wmma_bf16(pa, vb, acc[j]);
    }
    __syncthreads();                             // protect K/V tiles for next iter
  }

  // normalize + write O as [B,S, h*64 + d] bf16
#pragma unroll
  for (int j = 0; j < 4; ++j)
#pragma unroll
    for (int r = 0; r < 8; ++r) {
      const int tok = b * SEQ + qt * 16 + r + hi8;
      const float val = acc[j][r] / lrow[r];
      O[(size_t)tok * EMBED + h * HD + j * 16 + m] = (__bf16)val;
    }
}

// ---------------- Output projection ----------------
// C[32x64] per wave, fp32 output + bias.
__global__ void out_gemm_kernel(const bf16_t* __restrict__ X,
                                const bf16_t* __restrict__ W,
                                const float* __restrict__ bias,
                                float* __restrict__ out) {
  const int lane = threadIdx.x & 31;
  const int wave = threadIdx.x >> 5;
  const int gw   = blockIdx.x * (blockDim.x >> 5) + wave;
  const int CT   = EMBED / 64;                   // 16 col tiles
  const int rt   = gw / CT;                      // 0..255 (32-row tiles)
  const int ct   = gw % CT;
  if (rt >= NTOK / 32) return;

  const int m    = lane & 15;
  const int hi16 = (lane >> 4) << 4;
  const int hi8  = (lane >> 4) << 3;
  const int oc0  = ct * 64;

  const bf16_t* xrow0 = X + (size_t)(rt * 32 + m) * EMBED;
  const bf16_t* xrow1 = xrow0 + (size_t)16 * EMBED;

  v8f acc[2][4];
#pragma unroll
  for (int g = 0; g < 2; ++g)
#pragma unroll
    for (int j = 0; j < 4; ++j) acc[g][j] = (v8f){0.f,0.f,0.f,0.f,0.f,0.f,0.f,0.f};

  for (int k0 = 0; k0 < EMBED; k0 += 32) {
    __builtin_prefetch(xrow0 + k0 + 64, 0, 3);
    __builtin_prefetch(xrow1 + k0 + 64, 0, 3);
    const v16bf a0 = load_a_frag(xrow0, k0, lane);
    const v16bf a1 = load_a_frag(xrow1, k0, lane);
    v16bf b[4];
#pragma unroll
    for (int j = 0; j < 4; ++j)
      b[j] = load_b_frag(W + (size_t)(oc0 + j * 16 + m) * EMBED + k0 + hi16);
#pragma unroll
    for (int j = 0; j < 4; ++j) {
      acc[0][j] = wmma_bf16(a0, b[j], acc[0][j]);
      acc[1][j] = wmma_bf16(a1, b[j], acc[1][j]);
    }
  }

#pragma unroll
  for (int g = 0; g < 2; ++g)
#pragma unroll
    for (int j = 0; j < 4; ++j)
#pragma unroll
      for (int r = 0; r < 8; ++r) {
        const int tok = rt * 32 + g * 16 + r + hi8;
        const int o   = oc0 + j * 16 + m;
        out[(size_t)tok * EMBED + o] = acc[g][j][r] + bias[o];
      }
}

extern "C" void kernel_launch(void* const* d_in, const int* in_sizes, int n_in,
                              void* d_out, int out_size, void* d_ws, size_t ws_size,
                              hipStream_t stream) {
  const float* query = (const float*)d_in[0];
  // d_in[1] (key), d_in[2] (value) unused by the reference math
  const float* W_qkv = (const float*)d_in[3];
  const float* b_qkv = (const float*)d_in[4];
  const float* W_out = (const float*)d_in[5];
  const float* b_out = (const float*)d_in[6];

  char* ws = (char*)d_ws;
  bf16_t* Xbf    = (bf16_t*)ws; ws += (size_t)NTOK * EMBED * 2;        // 16 MB
  bf16_t* Wqkvbf = (bf16_t*)ws; ws += (size_t)THREE_D * EMBED * 2;     //  6 MB
  bf16_t* Woutbf = (bf16_t*)ws; ws += (size_t)EMBED * EMBED * 2;       //  2 MB
  bf16_t* Qb     = (bf16_t*)ws; ws += (size_t)NTOK * EMBED * 2;        // 16 MB
  bf16_t* Kb     = (bf16_t*)ws; ws += (size_t)NTOK * EMBED * 2;        // 16 MB
  bf16_t* Vtb    = (bf16_t*)ws; ws += (size_t)NTOK * EMBED * 2;        // 16 MB
  bf16_t* Obf    = (bf16_t*)ws; ws += (size_t)NTOK * EMBED * 2;        // 16 MB

  cvt_bf16_kernel<<<2048, 256, 0, stream>>>(query, Xbf, NTOK * EMBED);
  cvt_bf16_kernel<<<2048, 256, 0, stream>>>(W_qkv, Wqkvbf, THREE_D * EMBED);
  cvt_bf16_kernel<<<1024, 256, 0, stream>>>(W_out, Woutbf, EMBED * EMBED);

  // 256 row tiles * 48 col tiles = 12288 waves / 8 per block
  qkv_gemm_kernel<<<1536, 256, 0, stream>>>(Xbf, Wqkvbf, b_qkv, Qb, Kb, Vtb);

  // 64 (b,h) * 16 blocks, 8 q-tiles per block
  flash_attn_kernel<<<1024, 256, 0, stream>>>(Qb, Kb, Vtb, Obf);

  // 256 row tiles * 16 col tiles = 4096 waves / 8 per block
  out_gemm_kernel<<<512, 256, 0, stream>>>(Obf, Woutbf, b_out, (float*)d_out);

  (void)in_sizes; (void)n_in; (void)out_size; (void)ws_size;
}
